// Attention_188978561266
// MI455X (gfx1250) — matrix operand order repaired and checked by reading, not compiled, not run on hardware
//
#include <hip/hip_runtime.h>
#include <hip/hip_bf16.h>
#include <math.h>

typedef __attribute__((ext_vector_type(16))) __bf16 v16bf;
typedef __attribute__((ext_vector_type(8)))  float  v8f;
typedef __attribute__((ext_vector_type(2)))  float  v2f;
typedef __attribute__((ext_vector_type(4)))  unsigned int u32x4;
typedef __attribute__((ext_vector_type(8)))  int i32x8;
typedef __attribute__((ext_vector_type(4)))  int i32x4;

#define BATCH   4
#define SEQ     1024
#define DMODEL  1024
#define NHEAD   16
#define DHEAD   64
#define M_ROWS  (BATCH*SEQ)               /* 4096 */
#define N_QKV   (3*DMODEL)                /* 3072 */
#define QKV_ELEMS (BATCH*NHEAD*SEQ*DHEAD) /* 4194304 */
#define GEMM1_BLOCKS (32*24)              /* 768 */
#define EPS 1e-6f
#define BP_LD 264                         /* bf16 units per pair-row (2*128 + pad) */

// ---------------------------------------------------------------- WMMA wrappers
__device__ __forceinline__ v8f wmma_bf16(v16bf a, v16bf b, v8f c) {
  return __builtin_amdgcn_wmma_f32_16x16x32_bf16(false, a, false, b, (short)0, c, false, false);
}
__device__ __forceinline__ v8f wmma_f32k4(v2f a, v2f b, v8f c) {
  return __builtin_amdgcn_wmma_f32_16x16x4_f32(false, a, false, b, (short)0, c, false, false);
}

// A fragment: 16x32 bf16 from row-major LDS tile base[m*ld + k]
__device__ __forceinline__ v16bf load_a_frag_bf16(const __bf16* base, int ld, int lane) {
  const int mA = lane & 15, ha = lane >> 4;
  const __bf16* row = base + mA * ld;
  v16bf a;
#pragma unroll
  for (int j = 0; j < 4; ++j) { int k = 16*ha + 2*j;     a[2*j]   = row[k]; a[2*j+1]   = row[k+1]; }
#pragma unroll
  for (int j = 0; j < 4; ++j) { int k = 16*ha + 8 + 2*j; a[8+2*j] = row[k]; a[8+2*j+1] = row[k+1]; }
  return a;
}

// B fragment: 32x16 bf16 from pair-interleaved LDS: base[(k>>1)*BP_LD + 2n + (k&1)]
// lane l: n=l&15; half=l>>4; VGPR j holds K = 16*half + 2j{,+1} -> one contiguous b32 pair
__device__ __forceinline__ v16bf load_b_frag_pairs(const __bf16* base, int lane) {
  const int nB = lane & 15, hb = lane >> 4;
  v16bf b;
#pragma unroll
  for (int j = 0; j < 8; ++j) {
    const __bf16* p = base + (8*hb + j) * BP_LD + 2*nB;
    b[2*j] = p[0]; b[2*j+1] = p[1];
  }
  return b;
}

__device__ __forceinline__ float half16_max(float v) {
#pragma unroll
  for (int m = 1; m < 16; m <<= 1) v = fmaxf(v, __shfl_xor(v, m, 32));
  return v;
}
__device__ __forceinline__ float half16_sum(float v) {
#pragma unroll
  for (int m = 1; m < 16; m <<= 1) v += __shfl_xor(v, m, 32);
  return v;
}
__device__ __forceinline__ float wave_sum(float v) {
#pragma unroll
  for (int m = 1; m < 32; m <<= 1) v += __shfl_xor(v, m, 32);
  return v;
}

// ---------------------------------------------------------------- TDM: K chunk -> LDS
// 2D tile: 64 f32 x 32 rows, row stride 64 f32; LDS padded by TDM to 65-dword rows
// (pad_enable=1, interval code 5 = 64 DWORDs, amount code 0 = 1 DWORD).
__device__ __forceinline__ void tdm_load_k_chunk(const float* gsrc, unsigned int lds_off) {
  const unsigned long long ga = (unsigned long long)(uintptr_t)gsrc;
  u32x4 g0;
  g0[0] = 1u;                                            // count=1, no gather
  g0[1] = lds_off;                                       // lds_addr (bytes)
  g0[2] = (unsigned int)ga;                              // global_addr[31:0]
  g0[3] = (unsigned int)((ga >> 32) & 0x1FFFFFFull)      // global_addr[56:32]
        | (2u << 30);                                    // type = 2 (image)
  i32x8 g1;
  g1[0] = (2 << 16) | (1 << 20) | (5 << 22);             // data_size=4B, pad_en, pad_interval=64DW, pad_amount=1DW
  g1[1] = (64 << 16);                                    // tensor_dim0 = 64   (bits 79:48)
  g1[2] = (1024 << 16);                                  // tensor_dim1 = 1024 (bits 111:80)
  g1[3] = (64 << 16);                                    // tile_dim0 = 64     (bits 127:112)
  g1[4] = 32;                                            // tile_dim1 = 32, tile_dim2 = 0
  g1[5] = 64;                                            // tensor_dim0_stride = 64
  g1[6] = 0;
  g1[7] = 0;
  i32x4 z4 = {0, 0, 0, 0};                               // groups 2/3 unused (2D)
  i32x8 z8 = {0, 0, 0, 0, 0, 0, 0, 0};                   // 6-arg toolchain extra group
  __builtin_amdgcn_tensor_load_to_lds(g0, g1, z4, z4, z8, 0);
}

// ---------------------------------------------------------------- kernel 1: qkv GEMM
__global__ void __launch_bounds__(256) qkv_gemm_kernel(
    const float* __restrict__ x, const float* __restrict__ Wqkv,
    float* __restrict__ qb, float* __restrict__ kb, float* __restrict__ vb,
    float* __restrict__ partq, float* __restrict__ partk)
{
  __shared__ __bf16 As[128][36];
  __shared__ __bf16 Bp[16][BP_LD];   // pair-interleaved B tile
  __shared__ float  red[16];

  const int tid  = threadIdx.x;
  const int lane = tid & 31, wid = tid >> 5;
  const int m0 = blockIdx.x * 128;
  const int n0 = blockIdx.y * 128;
  const int wr = wid >> 2, wc = wid & 3;   // 2x4 wave grid; wave tile = 64x32

  v8f acc[4][2] = {};

  const int amm = tid >> 1,  ah   = (tid & 1) * 16;   // A coop: 16 floats/thread
  const int bk2 = tid >> 4,  bns  = (tid & 15) * 8;   // B coop: 2 k-rows x 8 n/thread

  for (int k0 = 0; k0 < DMODEL; k0 += 32) {
    {
      const float4* sa = (const float4*)(x + (size_t)(m0 + amm) * DMODEL + k0 + ah);
#pragma unroll
      for (int i = 0; i < 4; ++i) {
        float4 v = sa[i];
        As[amm][ah+4*i+0] = (__bf16)v.x; As[amm][ah+4*i+1] = (__bf16)v.y;
        As[amm][ah+4*i+2] = (__bf16)v.z; As[amm][ah+4*i+3] = (__bf16)v.w;
      }
      const float4* s0 = (const float4*)(Wqkv + (size_t)(k0 + 2*bk2)     * N_QKV + n0 + bns);
      const float4* s1 = (const float4*)(Wqkv + (size_t)(k0 + 2*bk2 + 1) * N_QKV + n0 + bns);
      float4 r0a = s0[0], r0b = s0[1], r1a = s1[0], r1b = s1[1];
      __bf16* dst = &Bp[bk2][2*bns];
      dst[0]=(__bf16)r0a.x; dst[1]=(__bf16)r1a.x;  dst[2]=(__bf16)r0a.y; dst[3]=(__bf16)r1a.y;
      dst[4]=(__bf16)r0a.z; dst[5]=(__bf16)r1a.z;  dst[6]=(__bf16)r0a.w; dst[7]=(__bf16)r1a.w;
      dst[8]=(__bf16)r0b.x; dst[9]=(__bf16)r1b.x;  dst[10]=(__bf16)r0b.y; dst[11]=(__bf16)r1b.y;
      dst[12]=(__bf16)r0b.z; dst[13]=(__bf16)r1b.z; dst[14]=(__bf16)r0b.w; dst[15]=(__bf16)r1b.w;
      if (k0 + 32 < DMODEL) {
        __builtin_prefetch(x    + (size_t)(m0 + amm) * DMODEL + k0 + 32 + ah, 0, 0);
        __builtin_prefetch(Wqkv + (size_t)(k0 + 32 + 2*bk2) * N_QKV + n0 + bns, 0, 0);
      }
    }
    __syncthreads();

    v16bf af[4], bfr[2];
#pragma unroll
    for (int mt = 0; mt < 4; ++mt) af[mt]  = load_a_frag_bf16(&As[wr*64 + mt*16][0], 36, lane);
#pragma unroll
    for (int nt = 0; nt < 2; ++nt) bfr[nt] = load_b_frag_pairs(&Bp[0][2*(wc*32 + nt*16)], lane);
#pragma unroll
    for (int mt = 0; mt < 4; ++mt)
#pragma unroll
      for (int nt = 0; nt < 2; ++nt)
        acc[mt][nt] = wmma_bf16(af[mt], bfr[nt], acc[mt][nt]);
    __syncthreads();
  }

  float ssq = 0.f, ssk = 0.f;
  const int ha = lane >> 4, nB = lane & 15;
#pragma unroll
  for (int mt = 0; mt < 4; ++mt) {
#pragma unroll
    for (int nt = 0; nt < 2; ++nt) {
#pragma unroll
      for (int r = 0; r < 8; ++r) {
        int m = m0 + wr*64 + mt*16 + r + 8*ha;
        int n = n0 + wc*32 + nt*16 + nB;
        float v = acc[mt][nt][r];
        int which = n >> 10;
        int hh = (n >> 6) & 15, dd = n & 63;
        int b = m >> 10, s = m & 1023;
        int dst = ((b * NHEAD + hh) * SEQ + s) * DHEAD + dd;
        if (which == 0)      { qb[dst] = v; ssq += v * v; }
        else if (which == 1) { kb[dst] = v; ssk += v * v; }
        else                 { vb[dst] = v; }
      }
    }
  }
  ssq = wave_sum(ssq); ssk = wave_sum(ssk);
  if (lane == 0) { red[wid] = ssq; red[8 + wid] = ssk; }
  __syncthreads();
  if (tid == 0) {
    float tq = 0.f, tk = 0.f;
#pragma unroll
    for (int w = 0; w < 8; ++w) { tq += red[w]; tk += red[8 + w]; }
    int bid = blockIdx.y * gridDim.x + blockIdx.x;
    partq[bid] = tq; partk[bid] = tk;
  }
}

// ---------------------------------------------------------------- kernel 2: global RMS
__global__ void rms_kernel(const float* __restrict__ partq, const float* __restrict__ partk,
                           float* __restrict__ stats)
{
  if (threadIdx.x == 0) {
    float sq = 0.f, sk = 0.f;
    for (int i = 0; i < GEMM1_BLOCKS; ++i) { sq += partq[i]; sk += partk[i]; }
    float rq = sqrtf(sq / (float)QKV_ELEMS);
    float rk = sqrtf(sk / (float)QKV_ELEMS);
    stats[0] = 1.f / (rq + EPS);
    stats[1] = 1.f / (rk + EPS);
  }
}

// ---------------------------------------------------------------- kernel 3: fused attention
// All q/k normalization folded into Q registers; K streamed RAW by the TDM
// (double-buffered tensor_load_to_lds), V loaded cooperatively (transpose+bf16).
__global__ void __launch_bounds__(256) attention_kernel(
    const float* __restrict__ qb, const float* __restrict__ kb, const float* __restrict__ vb,
    const float* __restrict__ scale_q, const float* __restrict__ scale_k,
    const float* __restrict__ stats, float* __restrict__ z)
{
  __shared__ float  Ks[2][32][DHEAD + 1]; // raw K chunks, TDM-padded to 65-dword rows
  __shared__ __bf16 Vs[DHEAD][34];        // transposed V chunk [d][key], bf16
  __shared__ __bf16 Ps[8][16][36];        // per-wave P staging

  const int tid  = threadIdx.x;
  const int lane = tid & 31, wid = tid >> 5;
  const int bh = blockIdx.x;
  const int b  = bh >> 4, h = bh & 15;
  const int qtile = blockIdx.y;

  const float* Q = qb + (size_t)bh * SEQ * DHEAD;
  const float* K = kb + (size_t)bh * SEQ * DHEAD;
  const float* V = vb + (size_t)bh * SEQ * DHEAD;

  const int mA = lane & 15, ha = lane >> 4;
  const int nB = mA;

  // fold inv_q*inv_k*scale_q[d]*scale_k[d] entirely into Q
  const float gscale = stats[0] * stats[1];
  float qa[16][2];
  {
    const int srow = qtile * 128 + wid * 16 + mA;
    const float* qrow = Q + (size_t)srow * DHEAD;
#pragma unroll
    for (int c = 0; c < 16; ++c) {
      int d0 = 4*c + 2*ha;
      qa[c][0] = qrow[d0]     * gscale * scale_q[d0]     * scale_k[d0];
      qa[c][1] = qrow[d0 + 1] * gscale * scale_q[d0 + 1] * scale_k[d0 + 1];
    }
  }

  v8f o[4] = {};
  float mrun[8], lrun[8];
#pragma unroll
  for (int r = 0; r < 8; ++r) { mrun[r] = -1e30f; lrun[r] = 0.f; }

  const int ckk = tid >> 3, cds = (tid & 7) * 8;

  if (wid == 0)   // prime the pipe: chunk 0 -> buffer 0
    tdm_load_k_chunk(K, (unsigned int)(uintptr_t)&Ks[0][0][0]);

  for (int ic = 0; ic < 32; ++ic) {
    const int cur = ic & 1;
    __syncthreads();   // all waves done reading buffer cur (prev round) before TDM rewrites it
    if (wid == 0) {
      if (ic + 1 < 32) {
        tdm_load_k_chunk(K + (size_t)(ic + 1) * 32 * DHEAD,
                         (unsigned int)(uintptr_t)&Ks[(ic + 1) & 1][0][0]);
        __builtin_amdgcn_s_wait_tensorcnt((short)1);  // chunk ic resident
      } else {
        __builtin_amdgcn_s_wait_tensorcnt((short)0);
      }
    }
    {
      const float* vrow = V + (size_t)(ic * 32 + ckk) * DHEAD + cds;
#pragma unroll
      for (int i = 0; i < 8; ++i) Vs[cds + i][ckk] = (__bf16)vrow[i];
    }
    __syncthreads();

    // S (16 x 32) = Qn . K^T, fp32 WMMA
    v8f s0 = {}, s1 = {};
#pragma unroll
    for (int c = 0; c < 16; ++c) {
      v2f a; a[0] = qa[c][0]; a[1] = qa[c][1];
      int dk = 4*c + 2*ha;
      v2f b0; b0[0] = Ks[cur][nB][dk];      b0[1] = Ks[cur][nB][dk + 1];
      v2f b1; b1[0] = Ks[cur][16 + nB][dk]; b1[1] = Ks[cur][16 + nB][dk + 1];
      s0 = wmma_f32k4(a, b0, s0);
      s1 = wmma_f32k4(a, b1, s1);
    }

    // online softmax
    float p0[8], p1[8];
#pragma unroll
    for (int r = 0; r < 8; ++r) {
      float mx   = half16_max(fmaxf(s0[r], s1[r]));
      float mnew = fmaxf(mrun[r], mx);
      float alpha = __expf(mrun[r] - mnew);
      p0[r] = __expf(s0[r] - mnew);
      p1[r] = __expf(s1[r] - mnew);
      float rs = half16_sum(p0[r] + p1[r]);
      lrun[r] = lrun[r] * alpha + rs;
      mrun[r] = mnew;
#pragma unroll
      for (int t = 0; t < 4; ++t) o[t][r] *= alpha;
    }

    // stage P then reload as A fragment (same-wave LDS RAW -> drain DS counter)
#pragma unroll
    for (int r = 0; r < 8; ++r) {
      Ps[wid][r + 8*ha][nB]      = (__bf16)p0[r];
      Ps[wid][r + 8*ha][16 + nB] = (__bf16)p1[r];
    }
    asm volatile("s_wait_dscnt 0" ::: "memory");

    v16bf pa = load_a_frag_bf16(&Ps[wid][0][0], 36, lane);
#pragma unroll
    for (int t = 0; t < 4; ++t) {
      const __bf16* vr = &Vs[t*16 + nB][0];
      v16bf vf;
#pragma unroll
      for (int j = 0; j < 8; ++j) { int k = 16*ha + 2*j; vf[2*j] = vr[k]; vf[2*j+1] = vr[k+1]; }
      o[t] = wmma_bf16(pa, vf, o[t]);
    }
  }

  // write z as [b][s][h*64+d]
#pragma unroll
  for (int t = 0; t < 4; ++t) {
#pragma unroll
    for (int r = 0; r < 8; ++r) {
      int s = qtile * 128 + wid * 16 + r + 8*ha;
      int d = t*16 + nB;
      z[((size_t)(b * SEQ + s) * NHEAD + h) * DHEAD + d] = o[t][r] / lrun[r];
    }
  }
}

// ---------------------------------------------------------------- kernel 4: output projection
__global__ void __launch_bounds__(256) out_gemm_kernel(
    const float* __restrict__ zin, const float* __restrict__ Wo, float* __restrict__ out)
{
  __shared__ __bf16 As[128][36];
  __shared__ __bf16 Bp[16][BP_LD];

  const int tid  = threadIdx.x;
  const int lane = tid & 31, wid = tid >> 5;
  const int m0 = blockIdx.x * 128;
  const int n0 = blockIdx.y * 128;
  const int wr = wid >> 2, wc = wid & 3;

  v8f acc[4][2] = {};

  const int amm = tid >> 1, ah  = (tid & 1) * 16;
  const int bk2 = tid >> 4, bns = (tid & 15) * 8;

  for (int k0 = 0; k0 < DMODEL; k0 += 32) {
    {
      const float4* sa = (const float4*)(zin + (size_t)(m0 + amm) * DMODEL + k0 + ah);
#pragma unroll
      for (int i = 0; i < 4; ++i) {
        float4 v = sa[i];
        As[amm][ah+4*i+0] = (__bf16)v.x; As[amm][ah+4*i+1] = (__bf16)v.y;
        As[amm][ah+4*i+2] = (__bf16)v.z; As[amm][ah+4*i+3] = (__bf16)v.w;
      }
      const float4* s0 = (const float4*)(Wo + (size_t)(k0 + 2*bk2)     * DMODEL + n0 + bns);
      const float4* s1 = (const float4*)(Wo + (size_t)(k0 + 2*bk2 + 1) * DMODEL + n0 + bns);
      float4 r0a = s0[0], r0b = s0[1], r1a = s1[0], r1b = s1[1];
      __bf16* dst = &Bp[bk2][2*bns];
      dst[0]=(__bf16)r0a.x; dst[1]=(__bf16)r1a.x;  dst[2]=(__bf16)r0a.y; dst[3]=(__bf16)r1a.y;
      dst[4]=(__bf16)r0a.z; dst[5]=(__bf16)r1a.z;  dst[6]=(__bf16)r0a.w; dst[7]=(__bf16)r1a.w;
      dst[8]=(__bf16)r0b.x; dst[9]=(__bf16)r1b.x;  dst[10]=(__bf16)r0b.y; dst[11]=(__bf16)r1b.y;
      dst[12]=(__bf16)r0b.z; dst[13]=(__bf16)r1b.z; dst[14]=(__bf16)r0b.w; dst[15]=(__bf16)r1b.w;
      if (k0 + 32 < DMODEL) {
        __builtin_prefetch(zin + (size_t)(m0 + amm) * DMODEL + k0 + 32 + ah, 0, 0);
        __builtin_prefetch(Wo  + (size_t)(k0 + 32 + 2*bk2) * DMODEL + n0 + bns, 0, 0);
      }
    }
    __syncthreads();

    v16bf af[4], bfr[2];
#pragma unroll
    for (int mt = 0; mt < 4; ++mt) af[mt]  = load_a_frag_bf16(&As[wr*64 + mt*16][0], 36, lane);
#pragma unroll
    for (int nt = 0; nt < 2; ++nt) bfr[nt] = load_b_frag_pairs(&Bp[0][2*(wc*32 + nt*16)], lane);
#pragma unroll
    for (int mt = 0; mt < 4; ++mt)
#pragma unroll
      for (int nt = 0; nt < 2; ++nt)
        acc[mt][nt] = wmma_bf16(af[mt], bfr[nt], acc[mt][nt]);
    __syncthreads();
  }

  const int ha = lane >> 4, nB = lane & 15;
#pragma unroll
  for (int mt = 0; mt < 4; ++mt)
#pragma unroll
    for (int nt = 0; nt < 2; ++nt)
#pragma unroll
      for (int r = 0; r < 8; ++r) {
        int m = m0 + wr*64 + mt*16 + r + 8*ha;
        int n = n0 + wc*32 + nt*16 + nB;
        out[(size_t)m * DMODEL + n] = acc[mt][nt][r];
      }
}

// ---------------------------------------------------------------- launch
extern "C" void kernel_launch(void* const* d_in, const int* in_sizes, int n_in,
                              void* d_out, int out_size, void* d_ws, size_t ws_size,
                              hipStream_t stream)
{
  (void)in_sizes; (void)n_in; (void)out_size; (void)ws_size;
  const float* x    = (const float*)d_in[0];
  const float* Wqkv = (const float*)d_in[1];
  const float* Wo   = (const float*)d_in[2];
  const float* sq   = (const float*)d_in[3];
  const float* sk   = (const float*)d_in[4];
  float* out = (float*)d_out;

  float* qb    = (float*)d_ws;                 // [4*16][1024][64]
  float* kb    = qb + QKV_ELEMS;
  float* vb    = kb + QKV_ELEMS;
  float* zb    = vb + QKV_ELEMS;               // [4096][1024]
  float* partq = zb + (size_t)M_ROWS * DMODEL; // [768]
  float* partk = partq + GEMM1_BLOCKS;         // [768]
  float* stats = partk + GEMM1_BLOCKS;         // [2]

  qkv_gemm_kernel<<<dim3(32, 24), 256, 0, stream>>>(x, Wqkv, qb, kb, vb, partq, partk);
  rms_kernel<<<1, 32, 0, stream>>>(partq, partk, stats);
  attention_kernel<<<dim3(64, 8), 256, 0, stream>>>(qb, kb, vb, sq, sk, stats, zb);
  out_gemm_kernel<<<dim3(32, 8), 256, 0, stream>>>(zb, Wo, out);
}
